// OscillatorAttention_29892972380455
// MI455X (gfx1250) — matrix operand order repaired and checked by reading, not compiled, not run on hardware
//
#include <hip/hip_runtime.h>
#include <hip/hip_bf16.h>
#include <math.h>

// ---------------------------------------------------------------------------
// OscillatorAttention for MI455X (gfx1250, wave32, WMMA).
// Every matmul stage runs on v_wmma_f32_16x16x32_f16 (f16 in, f32 accum).
// Sparse top-k attention matrices are materialized dense (f16) so the
// propagation / attn-V stages ride the matrix cores instead of VALU gathers.
// Waves are register-blocked MTxNT so fragment loads amortize across several
// WMMA issues: operand traffic drops from 2KB/wmma to (1/NT + 1/MT) KB/wmma.
// ---------------------------------------------------------------------------

typedef __attribute__((ext_vector_type(16))) _Float16 v16h;
typedef __attribute__((ext_vector_type(8)))  _Float16 v8h;
typedef __attribute__((ext_vector_type(8)))  float    v8f;

#define DEV_INLINE __device__ __forceinline__

namespace osc {

constexpr int B  = 2;
constexpr int T  = 1024;
constexpr int DM = 1024;
constexpr int H  = 16;
constexpr int DK = 64;
constexpr int BH = B * H;        // 32 heads total
constexpr int TOPK = 128;

// ---------------------------------------------------------------------------
// Register-blocked WMMA core: C[mi][ni] += A(16xK @ m0+16mi) * Bt(16xK @
// n0+16ni)^T, both operands row-major f16.
// A-fragment (16-bit A 16x32, same k = 16*hi + s order as B): lane lo = row, hi selects K-halves
//   -> two 16B loads. B-fragment (32x16): lane lo = col, hi -> K base 16*hi,
//   16 contiguous K values -> one 32B load from row n of Bt.
// ---------------------------------------------------------------------------
template <int MT, int NT>
DEV_INLINE void wmma_block_f16(const _Float16* __restrict__ A, int lda,
                               const _Float16* __restrict__ Bt, int ldb,
                               int m0, int n0, int K, int lane,
                               v8f (&c)[MT][NT]) {
  const int lo = lane & 15, hi = lane >> 4;
  const _Float16* ap[MT];
  const _Float16* bp[NT];
  #pragma unroll
  for (int mi = 0; mi < MT; ++mi)
    ap[mi] = A + (size_t)(m0 + mi * 16 + lo) * lda + hi * 16;
  #pragma unroll
  for (int ni = 0; ni < NT; ++ni)
    bp[ni] = Bt + (size_t)(n0 + ni * 16 + lo) * ldb + hi * 16;

  #pragma unroll 2
  for (int k0 = 0; k0 < K; k0 += 32) {
    v16h a[MT], b[NT];
    #pragma unroll
    for (int mi = 0; mi < MT; ++mi) {
      v8h a0 = *(const v8h*)(ap[mi] + k0);
      v8h a1 = *(const v8h*)(ap[mi] + k0 + 8);
      #pragma unroll
      for (int i = 0; i < 8; ++i) { a[mi][i] = a0[i]; a[mi][i + 8] = a1[i]; }
    }
    #pragma unroll
    for (int ni = 0; ni < NT; ++ni) b[ni] = *(const v16h*)(bp[ni] + k0);
    #pragma unroll
    for (int mi = 0; mi < MT; ++mi)
      #pragma unroll
      for (int ni = 0; ni < NT; ++ni)
        // 8 args: (neg_a, A, neg_b, B, c_mod, C, reuse_a, reuse_b)
        c[mi][ni] = __builtin_amdgcn_wmma_f32_16x16x32_f16(
            false, a[mi], false, b[ni], (short)0, c[mi][ni], false, false);
  }
}

// C/D layout: lane lo = N column; vgpr v = row M = v + 8*hi.
DEV_INLINE int cd_m(int lane, int v) { return ((lane >> 4) << 3) + v; }
DEV_INLINE int cd_n(int lane)        { return lane & 15; }

// ---------------------------------------------------------------------------
// f32 -> f16 convert
// ---------------------------------------------------------------------------
__global__ void __launch_bounds__(256)
cvt_f32_f16_kernel(const float* __restrict__ in, _Float16* __restrict__ out, int n) {
  int i = blockIdx.x * blockDim.x + threadIdx.x;
  if (i < n) out[i] = (_Float16)in[i];
}

// ---------------------------------------------------------------------------
// Projection GEMM: (2048x1024) x (1024x1024)^T, MT=2 NT=4 (32x64 per wave).
// MODE 0: f16 head row-major (b,h,t,dk)          [Q, K]
// MODE 1: f16 head transposed (b,h,dk,t)         [V]
// MODE 2: f32 head row-major + f16 head transposed  [D_re, D_im]
// MODE 3: f32 plain (m,n) into d_out             [final W_O projection]
// ---------------------------------------------------------------------------
template <int MODE>
__global__ void __launch_bounds__(128)
proj_kernel(const _Float16* __restrict__ X, const _Float16* __restrict__ Wt,
            _Float16* __restrict__ O16, float* __restrict__ O32) {
  constexpr int MT = 2, NT = 4;
  const int wave = blockIdx.x * (blockDim.x >> 5) + (threadIdx.x >> 5);
  const int lane = threadIdx.x & 31;
  const int bm = wave >> 4;        // 64 block-rows of 32
  const int bn = wave & 15;        // 16 block-cols of 64
  const int m0 = bm * 32, n0 = bn * 64;
  v8f c[MT][NT] = {};
  wmma_block_f16<MT, NT>(X, DM, Wt, DM, m0, n0, DM, lane, c);
  #pragma unroll
  for (int mi = 0; mi < MT; ++mi) {
    #pragma unroll
    for (int ni = 0; ni < NT; ++ni) {
      #pragma unroll
      for (int v = 0; v < 8; ++v) {
        const int m = m0 + mi * 16 + cd_m(lane, v);
        const int n = n0 + ni * 16 + cd_n(lane);
        const int b = m >> 10, t = m & (T - 1);
        const int h = n >> 6,  dk = n & (DK - 1);
        if constexpr (MODE == 0) {
          O16[(((size_t)(b * H + h) * T + t) * DK) + dk] = (_Float16)c[mi][ni][v];
        } else if constexpr (MODE == 1) {
          O16[(((size_t)(b * H + h) * DK + dk) * T) + t] = (_Float16)c[mi][ni][v];
        } else if constexpr (MODE == 2) {
          O32[(((size_t)(b * H + h) * T + t) * DK) + dk] = c[mi][ni][v];
          O16[(((size_t)(b * H + h) * DK + dk) * T) + t] = (_Float16)c[mi][ni][v];
        } else {
          O32[(size_t)m * DM + n] = c[mi][ni][v];
        }
      }
    }
  }
}

// ---------------------------------------------------------------------------
// Per-head similarity GEMM: S[bh] = scale * (A1*B1^T [+ A2*B2^T]); A/B are
// (T x DK) f16 row-major per head, K = 64. MT=2 NT=4.
// PAIRS=1: QK^T scores.  PAIRS=2: Ur·Urᵀ + Ui·Uiᵀ phase similarity.
// ---------------------------------------------------------------------------
template <int PAIRS>
__global__ void __launch_bounds__(128)
sim_kernel(const _Float16* __restrict__ A1, const _Float16* __restrict__ B1,
           const _Float16* __restrict__ A2, const _Float16* __restrict__ B2,
           float* __restrict__ Out, float scale) {
  constexpr int MT = 2, NT = 4;
  const int bh = blockIdx.y;
  const int wave = blockIdx.x * (blockDim.x >> 5) + (threadIdx.x >> 5);
  const int lane = threadIdx.x & 31;
  const int bm = wave >> 4, bn = wave & 15;   // 32 x 16 wave tiles
  const int m0 = bm * 32, n0 = bn * 64;
  const size_t ho = (size_t)bh * T * DK;
  v8f c[MT][NT] = {};
  wmma_block_f16<MT, NT>(A1 + ho, DK, B1 + ho, DK, m0, n0, DK, lane, c);
  if constexpr (PAIRS == 2)
    wmma_block_f16<MT, NT>(A2 + ho, DK, B2 + ho, DK, m0, n0, DK, lane, c);
  float* o = Out + (size_t)bh * T * T;
  #pragma unroll
  for (int mi = 0; mi < MT; ++mi)
    #pragma unroll
    for (int ni = 0; ni < NT; ++ni)
      #pragma unroll
      for (int v = 0; v < 8; ++v) {
        const int m = m0 + mi * 16 + cd_m(lane, v);
        const int n = n0 + ni * 16 + cd_n(lane);
        o[(size_t)m * T + n] = scale * c[mi][ni][v];
      }
}

// ---------------------------------------------------------------------------
// Row-wise top-k + softmax over kept entries, zeros elsewhere.
// One 256-thread workgroup per row (4 contiguous elements per thread, so
// global index order == (tid, j) order for lax.top_k-style tie breaking).
// Threshold found by 32-iteration bisection on the value range.
// ---------------------------------------------------------------------------
__global__ void __launch_bounds__(256)
topk_softmax_kernel(const float* __restrict__ S, _Float16* __restrict__ W16,
                    float* __restrict__ W32, int k) {
  const int row = blockIdx.x;
  const int tid = threadIdx.x;
  const float* s = S + (size_t)row * T;
  __shared__ float red[256];
  __shared__ int ired[256];

  float x0 = s[4 * tid + 0], x1 = s[4 * tid + 1];
  float x2 = s[4 * tid + 2], x3 = s[4 * tid + 3];

  // row max
  red[tid] = fmaxf(fmaxf(x0, x1), fmaxf(x2, x3));
  __syncthreads();
  for (int sft = 128; sft > 0; sft >>= 1) {
    if (tid < sft) red[tid] = fmaxf(red[tid], red[tid + sft]);
    __syncthreads();
  }
  const float rowmax = red[0];
  __syncthreads();
  // row min
  red[tid] = fminf(fminf(x0, x1), fminf(x2, x3));
  __syncthreads();
  for (int sft = 128; sft > 0; sft >>= 1) {
    if (tid < sft) red[tid] = fminf(red[tid], red[tid + sft]);
    __syncthreads();
  }
  const float rowmin = red[0];
  __syncthreads();

  // bisect: keep invariant count(x >= lo) >= k, count(x >= hi) < k
  float lo = rowmin, hi = rowmax + 1.0f;
  for (int it = 0; it < 32; ++it) {
    const float mid = 0.5f * (lo + hi);
    ired[tid] = (x0 >= mid) + (x1 >= mid) + (x2 >= mid) + (x3 >= mid);
    __syncthreads();
    for (int sft = 128; sft > 0; sft >>= 1) {
      if (tid < sft) ired[tid] += ired[tid + sft];
      __syncthreads();
    }
    const int cnt = ired[0];
    __syncthreads();
    if (cnt >= k) lo = mid; else hi = mid;
  }
  const float thr = lo;

  // index-ordered rank among elements >= thr (ties broken by lowest index)
  const int f0 = x0 >= thr, f1 = x1 >= thr, f2 = x2 >= thr, f3 = x3 >= thr;
  const int myc = f0 + f1 + f2 + f3;
  ired[tid] = myc;
  __syncthreads();
  for (int ofs = 1; ofs < 256; ofs <<= 1) {   // Hillis-Steele inclusive scan
    const int v = (tid >= ofs) ? ired[tid - ofs] : 0;
    __syncthreads();
    ired[tid] += v;
    __syncthreads();
  }
  const int excl = ired[tid] - myc;
  const int r0 = excl, r1 = r0 + f0, r2 = r1 + f1, r3 = r2 + f2;
  const int s0 = f0 && (r0 < k), s1 = f1 && (r1 < k);
  const int s2 = f2 && (r2 < k), s3 = f3 && (r3 < k);

  // softmax over the kept set (rowmax is always kept)
  const float e0 = s0 ? __expf(x0 - rowmax) : 0.f;
  const float e1 = s1 ? __expf(x1 - rowmax) : 0.f;
  const float e2 = s2 ? __expf(x2 - rowmax) : 0.f;
  const float e3 = s3 ? __expf(x3 - rowmax) : 0.f;
  red[tid] = e0 + e1 + e2 + e3;
  __syncthreads();
  for (int sft = 128; sft > 0; sft >>= 1) {
    if (tid < sft) red[tid] += red[tid + sft];
    __syncthreads();
  }
  const float inv = 1.0f / red[0];

  _Float16* w = W16 + (size_t)row * T;
  w[4 * tid + 0] = (_Float16)(e0 * inv);
  w[4 * tid + 1] = (_Float16)(e1 * inv);
  w[4 * tid + 2] = (_Float16)(e2 * inv);
  w[4 * tid + 3] = (_Float16)(e3 * inv);
  if (W32) {
    float* w2 = W32 + (size_t)row * T;
    w2[4 * tid + 0] = e0 * inv;
    w2[4 * tid + 1] = e1 * inv;
    w2[4 * tid + 2] = e2 * inv;
    w2[4 * tid + 3] = e3 * inv;
  }
}

// ---------------------------------------------------------------------------
// Topology propagation step (MT=2, NT=2):
//   Dnext = (1-lam)*Dcur + lam * (Aadj @ Dcur)  [+ alpha*D0 on last step]
// Aadj: (BH,T,T) f16 dense; Dt: (BH,DK,T) f16 transposed copy of Dcur;
// Dcur/D0/Dnext: (BH,T,DK) f32. Epilogue also emits the next transposed
// f16 copy so the next step's B-fragment loads stay contiguous.
// ---------------------------------------------------------------------------
__global__ void __launch_bounds__(128)
prop_step_kernel(const _Float16* __restrict__ Aadj, const _Float16* __restrict__ Dt,
                 const float* __restrict__ Dcur, const float* __restrict__ D0,
                 float* __restrict__ Dnext, _Float16* __restrict__ Dtnext,
                 const float* __restrict__ logit_lam,
                 const float* __restrict__ log_alpha) {
  constexpr int MT = 2, NT = 2;
  const int bh = blockIdx.y;
  const int wave = blockIdx.x * (blockDim.x >> 5) + (threadIdx.x >> 5);
  const int lane = threadIdx.x & 31;
  const int bm = wave >> 1, bn = wave & 1;   // 32 x 2 wave tiles
  const int m0 = bm * 32, n0 = bn * 32;
  v8f c[MT][NT] = {};
  wmma_block_f16<MT, NT>(Aadj + (size_t)bh * T * T, T,
                         Dt + (size_t)bh * DK * T, T, m0, n0, T, lane, c);
  const float lam = 1.0f / (1.0f + __expf(-logit_lam[0]));
  const float alpha = D0 ? (1.0f / (1.0f + __expf(-log_alpha[0]))) : 0.0f;
  const float* dc = Dcur + (size_t)bh * T * DK;
  const float* d0 = D0 ? (D0 + (size_t)bh * T * DK) : nullptr;
  float* dn = Dnext + (size_t)bh * T * DK;
  _Float16* dt = Dtnext + (size_t)bh * DK * T;
  #pragma unroll
  for (int mi = 0; mi < MT; ++mi)
    #pragma unroll
    for (int ni = 0; ni < NT; ++ni)
      #pragma unroll
      for (int v = 0; v < 8; ++v) {
        const int m = m0 + mi * 16 + cd_m(lane, v);
        const int n = n0 + ni * 16 + cd_n(lane);
        const size_t idx = (size_t)m * DK + n;
        float val = (1.0f - lam) * dc[idx] + lam * c[mi][ni][v];
        if (d0) val += alpha * d0[idx];
        dn[idx] = val;
        dt[(size_t)n * T + m] = (_Float16)val;
      }
}

// ---------------------------------------------------------------------------
// Unit-normalize complex oscillator states: U = D / max(|D|, 1e-6), f32 in,
// f16 row-major out (feeds phase-similarity WMMA as A and Bt).
// ---------------------------------------------------------------------------
__global__ void __launch_bounds__(256)
phase_norm_kernel(const float* __restrict__ Dr, const float* __restrict__ Di,
                  _Float16* __restrict__ Ur, _Float16* __restrict__ Ui) {
  const int tok = blockIdx.x * blockDim.x + threadIdx.x;
  if (tok >= BH * T) return;
  const float* dr = Dr + (size_t)tok * DK;
  const float* di = Di + (size_t)tok * DK;
  float ss = 0.f;
  #pragma unroll 8
  for (int i = 0; i < DK; ++i) ss += dr[i] * dr[i] + di[i] * di[i];
  const float inv = 1.0f / fmaxf(sqrtf(ss), 1e-6f);
  _Float16* ur = Ur + (size_t)tok * DK;
  _Float16* ui = Ui + (size_t)tok * DK;
  #pragma unroll 8
  for (int i = 0; i < DK; ++i) {
    ur[i] = (_Float16)(dr[i] * inv);
    ui[i] = (_Float16)(di[i] * inv);
  }
}

// ---------------------------------------------------------------------------
// out_head = phase_attn @ V (MT=2, NT=2), written f16 (b,t, h*64+dk)
// row-major so the final W_O projection is a plain 2048x1024x1024 WMMA GEMM.
// ---------------------------------------------------------------------------
__global__ void __launch_bounds__(128)
attnv_kernel(const _Float16* __restrict__ P, const _Float16* __restrict__ Vt,
             _Float16* __restrict__ OutH) {
  constexpr int MT = 2, NT = 2;
  const int bh = blockIdx.y;
  const int b = bh >> 4, h = bh & (H - 1);
  const int wave = blockIdx.x * (blockDim.x >> 5) + (threadIdx.x >> 5);
  const int lane = threadIdx.x & 31;
  const int bm = wave >> 1, bn = wave & 1;
  const int m0 = bm * 32, n0 = bn * 32;
  v8f c[MT][NT] = {};
  wmma_block_f16<MT, NT>(P + (size_t)bh * T * T, T,
                         Vt + (size_t)bh * DK * T, T, m0, n0, T, lane, c);
  #pragma unroll
  for (int mi = 0; mi < MT; ++mi)
    #pragma unroll
    for (int ni = 0; ni < NT; ++ni)
      #pragma unroll
      for (int v = 0; v < 8; ++v) {
        const int t = m0 + mi * 16 + cd_m(lane, v);
        const int dk = n0 + ni * 16 + cd_n(lane);
        OutH[((size_t)(b * T + t) * DM) + h * DK + dk] = (_Float16)c[mi][ni][v];
      }
}

}  // namespace osc

// ---------------------------------------------------------------------------
// Launch sequence. Workspace map (bytes, ~288 MB total with reuse):
//   0        xh (4MB) | 4M..16M  six f16 weights (2MB each)
//   16M Qh->Ur (4MB) | 20M Kh->Ui (4MB) | 24M Vt (4MB)
//   28M Dr0 | 36M Di0 | 44M DrB | 52M DiB | 60M DrC | 68M DiC  (8MB f32 each)
//   76M DrtA | 80M DrtB | 84M DitA | 88M DitB (4MB f16 each)
//   92M  adjacency/phase f16 dense (64MB, reused)
//   156M scores/phase_sim f32 dense (128MB, reused)
//   284M out_head f16 (4MB)
// ---------------------------------------------------------------------------
extern "C" void kernel_launch(void* const* d_in, const int* in_sizes, int n_in,
                              void* d_out, int out_size, void* d_ws, size_t ws_size,
                              hipStream_t stream) {
  using namespace osc;
  (void)in_sizes; (void)n_in; (void)out_size; (void)ws_size;

  const float* x   = (const float*)d_in[0];
  const float* WQ  = (const float*)d_in[1];
  const float* WK  = (const float*)d_in[2];
  const float* Wre = (const float*)d_in[3];
  const float* Wim = (const float*)d_in[4];
  const float* WV  = (const float*)d_in[5];
  const float* WO  = (const float*)d_in[6];
  const float* logit_lam = (const float*)d_in[7];
  const float* log_alpha = (const float*)d_in[8];

  float* out = (float*)d_out;
  float* phase_out = out + (size_t)B * T * DM;  // second tuple element

  char* w = (char*)d_ws;
  const size_t MB = (size_t)1 << 20;
  auto h16 = [&](size_t off) { return (_Float16*)(w + off); };
  auto f32p = [&](size_t off) { return (float*)(w + off); };

  _Float16* xh   = h16(0);
  _Float16* wqh  = h16(4 * MB);
  _Float16* wkh  = h16(6 * MB);
  _Float16* wreh = h16(8 * MB);
  _Float16* wimh = h16(10 * MB);
  _Float16* wvh  = h16(12 * MB);
  _Float16* woh  = h16(14 * MB);
  _Float16* Qh   = h16(16 * MB);   // reused as Ur
  _Float16* Kh   = h16(20 * MB);   // reused as Ui
  _Float16* Vth  = h16(24 * MB);
  float* Dr0 = f32p(28 * MB);
  float* Di0 = f32p(36 * MB);
  float* DrB = f32p(44 * MB);
  float* DiB = f32p(52 * MB);
  float* DrC = f32p(60 * MB);
  float* DiC = f32p(68 * MB);
  _Float16* DrtA = h16(76 * MB);
  _Float16* DrtB = h16(80 * MB);
  _Float16* DitA = h16(84 * MB);
  _Float16* DitB = h16(88 * MB);
  _Float16* Aadj = h16(92 * MB);   // reused as dense f16 phase_attn
  float* S32 = f32p(156 * MB);     // reused for phase_sim
  _Float16* outH = h16(284 * MB);

  const dim3 blk(128);
  const dim3 projGrid(256);             // 64x16 wave tiles (32x64 each)
  const dim3 simGrid(128, BH);          // 32x16 wave tiles per head
  const dim3 headGrid(16, BH);          // 32x2 wave tiles per head

  // 1) f32 -> f16 operand conversion
  const int nx = B * T * DM, nw = DM * DM;
  cvt_f32_f16_kernel<<<(nx + 255) / 256, 256, 0, stream>>>(x, xh, nx);
  cvt_f32_f16_kernel<<<(nw + 255) / 256, 256, 0, stream>>>(WQ, wqh, nw);
  cvt_f32_f16_kernel<<<(nw + 255) / 256, 256, 0, stream>>>(WK, wkh, nw);
  cvt_f32_f16_kernel<<<(nw + 255) / 256, 256, 0, stream>>>(Wre, wreh, nw);
  cvt_f32_f16_kernel<<<(nw + 255) / 256, 256, 0, stream>>>(Wim, wimh, nw);
  cvt_f32_f16_kernel<<<(nw + 255) / 256, 256, 0, stream>>>(WV, wvh, nw);
  cvt_f32_f16_kernel<<<(nw + 255) / 256, 256, 0, stream>>>(WO, woh, nw);

  // 2) projections (WMMA): Q,K row-major heads; V transposed; D_re/D_im both
  proj_kernel<0><<<projGrid, blk, 0, stream>>>(xh, wqh, Qh, nullptr);
  proj_kernel<0><<<projGrid, blk, 0, stream>>>(xh, wkh, Kh, nullptr);
  proj_kernel<1><<<projGrid, blk, 0, stream>>>(xh, wvh, Vth, nullptr);
  proj_kernel<2><<<projGrid, blk, 0, stream>>>(xh, wreh, DrtA, Dr0);
  proj_kernel<2><<<projGrid, blk, 0, stream>>>(xh, wimh, DitA, Di0);

  // 3) adjacency scores = QK^T / sqrt(dk)  (WMMA), then top-k softmax -> A
  sim_kernel<1><<<simGrid, blk, 0, stream>>>(Qh, Kh, nullptr, nullptr, S32, 0.125f);
  topk_softmax_kernel<<<BH * T, 256, 0, stream>>>(S32, Aadj, nullptr, TOPK);

  // 4) topology propagation, 4 steps x {re, im} (WMMA A@D, dense A)
  prop_step_kernel<<<headGrid, blk, 0, stream>>>(Aadj, DrtA, Dr0, nullptr, DrB, DrtB, logit_lam, log_alpha);
  prop_step_kernel<<<headGrid, blk, 0, stream>>>(Aadj, DrtB, DrB, nullptr, DrC, DrtA, logit_lam, log_alpha);
  prop_step_kernel<<<headGrid, blk, 0, stream>>>(Aadj, DrtA, DrC, nullptr, DrB, DrtB, logit_lam, log_alpha);
  prop_step_kernel<<<headGrid, blk, 0, stream>>>(Aadj, DrtB, DrB, Dr0,    DrC, DrtA, logit_lam, log_alpha);
  prop_step_kernel<<<headGrid, blk, 0, stream>>>(Aadj, DitA, Di0, nullptr, DiB, DitB, logit_lam, log_alpha);
  prop_step_kernel<<<headGrid, blk, 0, stream>>>(Aadj, DitB, DiB, nullptr, DiC, DitA, logit_lam, log_alpha);
  prop_step_kernel<<<headGrid, blk, 0, stream>>>(Aadj, DitA, DiC, nullptr, DiB, DitB, logit_lam, log_alpha);
  prop_step_kernel<<<headGrid, blk, 0, stream>>>(Aadj, DitB, DiB, Di0,    DiC, DitA, logit_lam, log_alpha);

  // 5) unit-normalize -> Ur/Ui (reuse Q/K buffers)
  phase_norm_kernel<<<(BH * T + 255) / 256, 256, 0, stream>>>(DrC, DiC, Qh, Kh);

  // 6) phase similarity = (UrUr^T + UiUi^T) * sqrt(2*dk)  (WMMA, 2 pairs)
  sim_kernel<2><<<simGrid, blk, 0, stream>>>(Qh, Qh, Kh, Kh, S32, 11.3137085f);

  // 7) phase top-k softmax -> dense f16 P (reuse Aadj) + f32 phase_attn output
  topk_softmax_kernel<<<BH * T, 256, 0, stream>>>(S32, Aadj, phase_out, TOPK);

  // 8) out = P @ V  (WMMA), assembled into (B,T,D_MODEL) f16
  attnv_kernel<<<headGrid, blk, 0, stream>>>(Aadj, Vth, outH);

  // 9) final projection out @ W_O^T  (WMMA), f32 straight into d_out
  proj_kernel<3><<<projGrid, blk, 0, stream>>>(outH, woh, nullptr, out);
}
